// Mamba2Mixer_80736795231101
// MI455X (gfx1250) — compile-verified, matrix operand order checked by reading, not run on hardware
//
#include <hip/hip_runtime.h>
#include <hip/hip_bf16.h>
#include <math.h>

#define HIDDEN    2560
#define NUM_HEADS 80
#define HEAD_DIM  64
#define STATE     128
#define CONV_K    4
#define INTER     (2*HIDDEN)              // 5120
#define CONV_DIM  (INTER + 2*STATE)       // 5376
#define PROJ      (INTER + CONV_DIM + NUM_HEADS) // 10576
#define SEQ       2048
#define BATCH     2
#define NTOK      (BATCH*SEQ)             // 4096
#define EPSF      1e-5f

typedef float  v8f   __attribute__((ext_vector_type(8)));
typedef float  v4f   __attribute__((ext_vector_type(4)));
typedef __bf16 v16bf __attribute__((ext_vector_type(16)));

// ---------------------------------------------------------------------------
// Load one 16x32 bf16 WMMA operand fragment from an fp32 K-major matrix.
// Caller guarantees `row` is in-bounds (clamped); out-of-range tile rows are
// clamped to a valid row — they only feed output rows/cols that the epilogue
// never stores, so no masking is needed in the hot loop.
// Per ISA layout: lanes 0-15 hold K {0..7,16..23}, lanes 16-31 hold {8..15,24..31}.
// ---------------------------------------------------------------------------
__device__ inline v16bf load_frag_nt(const float* __restrict__ p) {
    v16bf f;
    v4f a = *(const v4f*)(p);
    v4f b = *(const v4f*)(p + 4);
    v4f c = *(const v4f*)(p + 16);
    v4f d = *(const v4f*)(p + 20);
    f[0]  = (__bf16)a.x;  f[1]  = (__bf16)a.y;  f[2]  = (__bf16)a.z;  f[3]  = (__bf16)a.w;
    f[4]  = (__bf16)b.x;  f[5]  = (__bf16)b.y;  f[6]  = (__bf16)b.z;  f[7]  = (__bf16)b.w;
    f[8]  = (__bf16)c.x;  f[9]  = (__bf16)c.y;  f[10] = (__bf16)c.z;  f[11] = (__bf16)c.w;
    f[12] = (__bf16)d.x;  f[13] = (__bf16)d.y;  f[14] = (__bf16)d.z;  f[15] = (__bf16)d.w;
    return f;
}

// ---------------------------------------------------------------------------
// C[M,N] = A[M,K] * B[N,K]^T   (fp32 in/out, bf16 v_wmma compute, fp32 accum)
// Block: 256 threads = 8 waves in a 4(M) x 2(N) grid; block tile 128x128;
// each wave owns a 32x64 tile = 2x4 WMMA accumulators.
// ---------------------------------------------------------------------------
__global__ __launch_bounds__(256)
void gemm_nt_bf16_wmma(const float* __restrict__ A, const float* __restrict__ B,
                       float* __restrict__ C, int M, int N, int K) {
    const int lane = threadIdx.x & 31;
    const int wave = threadIdx.x >> 5;
    const int wm   = wave & 3;            // 0..3 along M
    const int wn   = wave >> 2;           // 0..1 along N
    const int m0   = blockIdx.y * 128 + wm * 32;
    const int n0   = blockIdx.x * 128 + wn * 64;
    const int lr   = lane & 15;
    const int khalf = (lane >> 4) << 3;   // 0 or 8

    // Loop-invariant, clamped base pointers (one per fragment).
    const float* ap[2];
    const float* bp[4];
    #pragma unroll
    for (int mt = 0; mt < 2; ++mt) {
        int row = m0 + mt * 16 + lr;
        row = row < M ? row : M - 1;
        ap[mt] = A + (long)row * K + khalf;
    }
    #pragma unroll
    for (int nt = 0; nt < 4; ++nt) {
        int row = n0 + nt * 16 + lr;
        row = row < N ? row : N - 1;
        bp[nt] = B + (long)row * K + khalf;
    }

    v8f acc[2][4];
    #pragma unroll
    for (int mt = 0; mt < 2; ++mt)
        #pragma unroll
        for (int nt = 0; nt < 4; ++nt)
            #pragma unroll
            for (int r = 0; r < 8; ++r) acc[mt][nt][r] = 0.0f;

    for (int k0 = 0; k0 < K; k0 += 32) {
        v16bf af[2], bf[4];
        #pragma unroll
        for (int mt = 0; mt < 2; ++mt) af[mt] = load_frag_nt(ap[mt] + k0);
        #pragma unroll
        for (int nt = 0; nt < 4; ++nt) bf[nt] = load_frag_nt(bp[nt] + k0);
        #pragma unroll
        for (int mt = 0; mt < 2; ++mt)
            #pragma unroll
            for (int nt = 0; nt < 4; ++nt)
                acc[mt][nt] = __builtin_amdgcn_wmma_f32_16x16x32_bf16(
                    false, af[mt], false, bf[nt], (short)0, acc[mt][nt], false, false);
    }

    // C/D layout: VGPR r -> row (r + 8*(lane>=16)), col (lane & 15)
    #pragma unroll
    for (int mt = 0; mt < 2; ++mt) {
        int rbase = m0 + mt * 16 + ((lane >> 4) << 3);
        #pragma unroll
        for (int nt = 0; nt < 4; ++nt) {
            int col = n0 + nt * 16 + lr;
            if (col < N) {
                #pragma unroll
                for (int r = 0; r < 8; ++r) {
                    int row = rbase + r;
                    if (row < M) C[(long)row * N + col] = acc[mt][nt][r];
                }
            }
        }
    }
}

// ---------------------------------------------------------------------------
// Causal depthwise conv1d (K=4) + bias + SiLU over the xBC slice of proj.
// ---------------------------------------------------------------------------
__global__ void conv_silu_kernel(const float* __restrict__ proj,
                                 const float* __restrict__ w,
                                 const float* __restrict__ bias,
                                 float* __restrict__ xBC) {
    long idx = (long)blockIdx.x * blockDim.x + threadIdx.x;
    const long total = (long)NTOK * CONV_DIM;
    if (idx >= total) return;
    int  ch = (int)(idx % CONV_DIM);
    long m  = idx / CONV_DIM;             // b*SEQ + s
    int  s  = (int)(m % SEQ);
    float acc = bias[ch];
    #pragma unroll
    for (int j = 0; j < CONV_K; ++j) {
        int sj = s - (CONV_K - 1) + j;
        if (sj >= 0)
            acc += w[ch * CONV_K + j] * proj[(m + (long)(sj - s)) * PROJ + INTER + ch];
    }
    float sig = 1.0f / (1.0f + __expf(-acc));
    xBC[idx] = acc * sig;
}

// ---------------------------------------------------------------------------
// dt = clip(softplus(proj_dt + dt_bias), 0, 100)
// ---------------------------------------------------------------------------
__global__ void dt_kernel(const float* __restrict__ proj,
                          const float* __restrict__ dt_bias,
                          float* __restrict__ dt) {
    long idx = (long)blockIdx.x * blockDim.x + threadIdx.x;
    const long total = (long)NTOK * NUM_HEADS;
    if (idx >= total) return;
    int  h = (int)(idx % NUM_HEADS);
    long m = idx / NUM_HEADS;
    float x  = proj[m * PROJ + INTER + CONV_DIM + h] + dt_bias[h];
    float sp = (x > 20.0f) ? x : log1pf(__expf(x));
    dt[idx] = fminf(fmaxf(sp, 0.0f), 100.0f);
}

// ---------------------------------------------------------------------------
// Exact SSM recurrence, one block per (batch, head).
// State S[64 p][128 n] lives in registers: thread (p = tid>>2, q = tid&3)
// owns n in [q*32, q*32+32).  B/C rows staged via LDS each step; the 4-lane
// y-reduction uses wave32 xor-shuffles.
//   S = S*exp(A*dt) + (x*dt) B ;  y = C.S + D*x
// ---------------------------------------------------------------------------
__global__ __launch_bounds__(256)
void ssm_scan_kernel(const float* __restrict__ xBC, const float* __restrict__ dt,
                     const float* __restrict__ A_log, const float* __restrict__ Dp,
                     float* __restrict__ y) {
    const int bh  = blockIdx.x;           // 0 .. BATCH*NUM_HEADS-1
    const int bi  = bh / NUM_HEADS;
    const int h   = bh % NUM_HEADS;
    const int tid = threadIdx.x;
    const int p   = tid >> 2;             // 0..63
    const int q   = tid & 3;              // 0..3
    const int n0  = q * 32;
    const float Ah = -__expf(A_log[h]);
    const float Dh = Dp[h];

    float S[32];
    #pragma unroll
    for (int i = 0; i < 32; ++i) S[i] = 0.0f;

    __shared__ float sB[STATE];
    __shared__ float sC[STATE];

    for (int t = 0; t < SEQ; ++t) {
        long m = (long)bi * SEQ + t;
        const float* row = xBC + m * CONV_DIM;
        if (tid < STATE)            sB[tid]         = row[INTER + tid];
        else if (tid < 2 * STATE)   sC[tid - STATE] = row[INTER + tid];
        __syncthreads();

        float dtv   = dt[m * NUM_HEADS + h];
        float decay = __expf(Ah * dtv);
        float xraw  = row[(long)h * HEAD_DIM + p];
        float xv    = xraw * dtv;

        float acc = 0.0f;
        #pragma unroll
        for (int i = 0; i < 32; ++i) {
            S[i] = S[i] * decay + xv * sB[n0 + i];
            acc += S[i] * sC[n0 + i];
        }
        acc += __shfl_xor(acc, 1, 32);
        acc += __shfl_xor(acc, 2, 32);
        if (q == 0)
            y[m * INTER + (long)h * HEAD_DIM + p] = acc + Dh * xraw;
        __syncthreads();
    }
}

// ---------------------------------------------------------------------------
// y = y_ssm * silu(gate); RMSNorm over INTER with norm_weight. One block/token.
// ---------------------------------------------------------------------------
__global__ __launch_bounds__(256)
void gate_norm_kernel(const float* __restrict__ yssm, const float* __restrict__ proj,
                      const float* __restrict__ nw, float* __restrict__ yn) {
    const long m   = blockIdx.x;
    const int  tid = threadIdx.x;
    constexpr int PT = INTER / 256;       // 20 elements per thread

    __shared__ float red[8];
    __shared__ float sscale;

    float vals[PT];
    float ss = 0.0f;
    #pragma unroll
    for (int i = 0; i < PT; ++i) {
        int c = tid + i * 256;
        float g = proj[m * PROJ + c];
        float v = yssm[m * INTER + c] * (g / (1.0f + __expf(-g)));
        vals[i] = v;
        ss += v * v;
    }
    #pragma unroll
    for (int o = 16; o > 0; o >>= 1) ss += __shfl_xor(ss, o, 32);
    if ((tid & 31) == 0) red[tid >> 5] = ss;
    __syncthreads();
    if (tid == 0) {
        float t = 0.0f;
        #pragma unroll
        for (int i = 0; i < 8; ++i) t += red[i];
        sscale = rsqrtf(t / (float)INTER + EPSF);
    }
    __syncthreads();
    float sc = sscale;
    #pragma unroll
    for (int i = 0; i < PT; ++i) {
        int c = tid + i * 256;
        yn[m * INTER + c] = vals[i] * sc * nw[c];
    }
}

// ---------------------------------------------------------------------------
extern "C" void kernel_launch(void* const* d_in, const int* in_sizes, int n_in,
                              void* d_out, int out_size, void* d_ws, size_t ws_size,
                              hipStream_t stream) {
    const float* x       = (const float*)d_in[0];   // (2,2048,2560)
    const float* w_in    = (const float*)d_in[1];   // (PROJ, HIDDEN)
    const float* conv_w  = (const float*)d_in[2];   // (CONV_DIM, 4)
    const float* conv_b  = (const float*)d_in[3];   // (CONV_DIM,)
    const float* A_log   = (const float*)d_in[4];   // (80,)
    const float* Dp      = (const float*)d_in[5];   // (80,)
    const float* dt_bias = (const float*)d_in[6];   // (80,)
    const float* norm_w  = (const float*)d_in[7];   // (INTER,)
    const float* w_out   = (const float*)d_in[8];   // (HIDDEN, INTER)
    float* out = (float*)d_out;
    float* ws  = (float*)d_ws;

    float* proj = ws;                                   // NTOK*PROJ     (173 MB)
    float* xBC  = proj + (size_t)NTOK * PROJ;           // NTOK*CONV_DIM ( 88 MB)
    float* dtb  = xBC  + (size_t)NTOK * CONV_DIM;       // NTOK*NUM_HEADS
    float* yssm = dtb  + (size_t)NTOK * NUM_HEADS;      // NTOK*INTER    ( 84 MB)
    float* yn   = xBC;                                  // reuse xBC after SSM

    // 1) proj = x @ in_proj_w^T  (M=4096, N=10576, K=2560)
    gemm_nt_bf16_wmma<<<dim3((PROJ + 127) / 128, NTOK / 128), 256, 0, stream>>>(
        x, w_in, proj, NTOK, PROJ, HIDDEN);

    // 2) causal depthwise conv + SiLU over xBC slice
    {
        long total = (long)NTOK * CONV_DIM;
        conv_silu_kernel<<<(unsigned)((total + 255) / 256), 256, 0, stream>>>(
            proj, conv_w, conv_b, xBC);
    }

    // 3) dt = clip(softplus(dt_raw + bias), 0, 100)
    {
        long total = (long)NTOK * NUM_HEADS;
        dt_kernel<<<(unsigned)((total + 255) / 256), 256, 0, stream>>>(
            proj, dt_bias, dtb);
    }

    // 4) SSM recurrence + D residual, one block per (batch, head)
    ssm_scan_kernel<<<BATCH * NUM_HEADS, 256, 0, stream>>>(xBC, dtb, A_log, Dp, yssm);

    // 5) gate * silu + RMSNorm
    gate_norm_kernel<<<NTOK, 256, 0, stream>>>(yssm, proj, norm_w, yn);

    // 6) out = yn @ out_proj_w^T  (M=4096, N=2560, K=5120)
    gemm_nt_bf16_wmma<<<dim3(HIDDEN / 128, NTOK / 128), 256, 0, stream>>>(
        yn, w_out, out, NTOK, HIDDEN, INTER);
}
